// EDWNet_74019466379555
// MI455X (gfx1250) — matrix operand order repaired and checked by reading, not compiled, not run on hardware
//
#include <hip/hip_runtime.h>
#include <hip/hip_bf16.h>

typedef __attribute__((ext_vector_type(16))) _Float16 v16h;
typedef __attribute__((ext_vector_type(8)))  float    v8f;
typedef __attribute__((ext_vector_type(4)))  float    f4;
typedef __attribute__((ext_vector_type(4)))  _Float16 h4;

#define N_NODES 30000
#define N_ADJ   400000
#define N_COL   200000
#define NODE_IN 46
#define EDGE_F  7
#define MID     64
#define DEPTH   6

// ---------------- generic gather descriptor: up to 3 concatenated segments ----
struct Gather3 {
    const float* ptr[3];
    const int*   idx[3];   // nullptr -> identity (row index)
    int          width[3]; // 0 -> unused
    int          ld[3];
};

// ---------------- fused gather + fp16 WMMA GEMM ------------------------------
// Tile: 128 rows x 64 cols per block (8 waves, each wave 16 rows x 64 cols).
// A and B tiles are staged in LDS directly in the WMMA fragment layout so the
// per-wave fragment load is a single 32B vector LDS read (2x ds_load_b128).
#define TM 128
#define TN 64
#define TK 32
#define FR_STRIDE 24   // halves per lane slot (48B, 16B aligned, bank-staggered)

__global__ __launch_bounds__(256)
void wmma_gemm_gather(const float* __restrict__ W, int ldw,
                      const float* __restrict__ bias,
                      float* __restrict__ Y, int ldy, int ycol0,
                      int M, int N, int K, Gather3 g)
{
    // fragment-layout staging: [tile][lane][16 halves used of FR_STRIDE]
    __shared__ _Float16 Afrag[8][32][FR_STRIDE] __attribute__((aligned(16)));
    __shared__ _Float16 Bfrag[4][32][FR_STRIDE] __attribute__((aligned(16)));

    const int tid  = threadIdx.x;
    const int lane = tid & 31;
    const int wave = tid >> 5;
    const int rowBase = blockIdx.x * TM;
    const int colBase = blockIdx.y * TN;

    v8f acc[4] = {};

    for (int k0 = 0; k0 < K; k0 += TK) {
        // ---- tail chunk: clear A staging (positions k>=K must be zero) ----
        if (k0 + TK > K) {
            for (int e = tid; e < 8 * 32 * 4; e += 256) {
                int wl = e >> 2, q = e & 3;
                *(unsigned long long*)&Afrag[wl >> 5][wl & 31][q * 4] = 0ull;
            }
            __syncthreads();
        }

        // ---- cooperative A-tile gather into fragment layout ----
        int koff = 0;
        #pragma unroll
        for (int s = 0; s < 3; ++s) {
            const int sw = g.width[s];
            if (sw > 0) {
                const float* __restrict__ sp = g.ptr[s];
                const int*   __restrict__ si = g.idx[s];
                const int    sld = g.ld[s];
                const bool use4 = (((sld | sw | koff) & 3) == 0);
                if (use4) {
                    // float4 path: quad of consecutive k -> same lane, contiguous idx
                    for (int e = tid; e < TM * TK / 4; e += 256) {
                        int row = e >> 3;
                        int kk  = (e & 7) * 4;
                        int krem = k0 + kk - koff;
                        if (krem < 0 || krem >= sw) continue;
                        int grow = rowBase + row;
                        if (grow >= M) continue;
                        int r2 = si ? si[grow] : grow;
                        f4 v = *(const f4*)(sp + (size_t)r2 * sld + krem);
                        h4 h = __builtin_convertvector(v, h4);
                        int fl  = (row & 15) + 16 * ((kk >> 3) & 1);
                        int idx = (kk & 7) + 8 * (kk >> 4);
                        *(h4*)&Afrag[row >> 4][fl][idx] = h;
                    }
                } else {
                    for (int e = tid; e < TM * TK; e += 256) {
                        int row = e >> 5;
                        int kk  = e & 31;
                        int krem = k0 + kk - koff;
                        if (krem < 0 || krem >= sw) continue;
                        int grow = rowBase + row;
                        if (grow >= M) continue;
                        int r2 = si ? si[grow] : grow;
                        float v = sp[(size_t)r2 * sld + krem];
                        int fl  = (row & 15) + 16 * ((kk >> 3) & 1);
                        int idx = (kk & 7) + 8 * (kk >> 4);
                        Afrag[row >> 4][fl][idx] = (_Float16)v;
                    }
                }
            }
            koff += sw;
        }

        // ---- B-tile (weights) into fragment layout, zero-padded ----
        for (int e = tid; e < TK * TN; e += 256) {
            int kk = e >> 6;
            int nn = e & 63;
            int k = k0 + kk;
            int col = colBase + nn;
            float v = (k < K && col < N) ? W[(size_t)k * ldw + col] : 0.f;
            int fl = (nn & 15) + 16 * ((kk >> 3) & 1);
            Bfrag[nn >> 4][fl][(kk & 7) + 8 * (kk >> 4)] = (_Float16)v;
        }
        __syncthreads();

        // ---- fragment loads are plain contiguous vector LDS reads ----
        const v16h a = *(const v16h*)(&Afrag[wave][lane][0]);
        #pragma unroll
        for (int s = 0; s < 4; ++s) {
            const v16h b = *(const v16h*)(&Bfrag[s][lane][0]);
            acc[s] = __builtin_amdgcn_wmma_f32_16x16x32_f16(
                false, a, false, b, (short)0, acc[s], false, false);
        }
        __syncthreads();
    }

    // ---- epilogue: bias add, f32 store (C/D layout: vgpr r -> m=r+8*half) ----
    #pragma unroll
    for (int s = 0; s < 4; ++s) {
        int col = colBase + s * 16 + (lane & 15);
        #pragma unroll
        for (int r = 0; r < 8; ++r) {
            int m = r + 8 * (lane >> 4);
            int grow = rowBase + wave * 16 + m;
            if (grow < M && col < N) {
                float v = acc[s][r];
                if (bias) v += bias[col];
                Y[(size_t)grow * ldy + ycol0 + col] = v;
            }
        }
    }
}

// ---------------- BatchNorm stats (64 columns): sum / sumsq reduction --------
__global__ __launch_bounds__(256)
void bn_stats64(const float* __restrict__ Y, int ld, int M, float* __restrict__ stats)
{
    int c   = threadIdx.x & 63;
    int grp = threadIdx.x >> 6; // 0..3
    float s = 0.f, sq = 0.f;
    for (int r = blockIdx.x * 4 + grp; r < M; r += gridDim.x * 4) {
        float v = Y[(size_t)r * ld + c];
        s += v; sq += v * v;
    }
    __shared__ float sh[2][4][64];
    sh[0][grp][c] = s; sh[1][grp][c] = sq;
    __syncthreads();
    if (grp == 0) {
        s  = sh[0][0][c] + sh[0][1][c] + sh[0][2][c] + sh[0][3][c];
        sq = sh[1][0][c] + sh[1][1][c] + sh[1][2][c] + sh[1][3][c];
        atomicAdd(&stats[c], s);
        atomicAdd(&stats[64 + c], sq);
    }
}

// mode 0: leaky-relu, 1: BN + leaky-relu, 2: sigmoid
__global__ __launch_bounds__(256)
void act_kernel(float* __restrict__ Y, int ld, int M, int C, int mode,
                const float* __restrict__ stats,
                const float* __restrict__ gamma, const float* __restrict__ beta,
                float invM)
{
    long long i = (long long)blockIdx.x * blockDim.x + threadIdx.x;
    long long total = (long long)M * C;
    if (i >= total) return;
    int r = (int)(i / C), c = (int)(i % C);
    float v = Y[(size_t)r * ld + c];
    if (mode == 1) {
        float mu  = stats[c] * invM;
        float var = stats[64 + c] * invM - mu * mu;
        v = gamma[c] * (v - mu) * rsqrtf(var + 1e-5f) + beta[c];
    }
    if (mode == 2) v = 1.f / (1.f + __expf(-v));
    else           v = v > 0.f ? v : 0.01f * v;
    Y[(size_t)r * ld + c] = v;
}

// ---------------- scatter-mean helpers ---------------------------------------
__global__ __launch_bounds__(256)
void fillk(float* __restrict__ p, long long n, float val)
{
    long long i = (long long)blockIdx.x * blockDim.x + threadIdx.x;
    if (i < n) p[i] = val;
}

__global__ __launch_bounds__(256)
void scatter_add64(const float* __restrict__ msg, const int* __restrict__ dst,
                   int E, float* __restrict__ agg)
{
    long long i = (long long)blockIdx.x * blockDim.x + threadIdx.x;
    if (i >= (long long)E * 64) return;
    int e = (int)(i >> 6), f = (int)(i & 63);
    atomicAdd(&agg[(size_t)dst[e] * 64 + f], msg[i]);
}

__global__ __launch_bounds__(256)
void scatter_cnt(const int* __restrict__ dst, int E, float* __restrict__ cnt)
{
    int e = blockIdx.x * blockDim.x + threadIdx.x;
    if (e < E) atomicAdd(&cnt[dst[e]], 1.f);
}

__global__ __launch_bounds__(256)
void divide64(float* __restrict__ agg, const float* __restrict__ cnt, int Nn)
{
    long long i = (long long)blockIdx.x * blockDim.x + threadIdx.x;
    if (i >= (long long)Nn * 64) return;
    agg[i] /= fmaxf(cnt[i >> 6], 1.f);
}

// gating multiply + optional residual, writes into FEATS slice (ld 448)
__global__ __launch_bounds__(256)
void gate_res(const float* __restrict__ b1, const float* __restrict__ b2,
              const float* __restrict__ resid, int ldr,
              float* __restrict__ outp, int ldo, int M)
{
    long long i = (long long)blockIdx.x * blockDim.x + threadIdx.x;
    if (i >= (long long)M * 64) return;
    int r = (int)(i >> 6), c = (int)(i & 63);
    float v = b1[i] * b2[i];
    if (resid) v += resid[(size_t)r * ldr + c];
    outp[(size_t)r * ldo + c] = v;
}

// final 64->1 linear + sigmoid
__global__ __launch_bounds__(256)
void final_head(const float* __restrict__ h, const float* __restrict__ w,
                const float* __restrict__ b, float* __restrict__ out, int M)
{
    __shared__ float ws[64];
    if (threadIdx.x < 64) ws[threadIdx.x] = w[threadIdx.x];
    __syncthreads();
    int r = blockIdx.x * blockDim.x + threadIdx.x;
    if (r >= M) return;
    float s = b[0];
    #pragma unroll 8
    for (int k = 0; k < 64; ++k) s += h[(size_t)r * 64 + k] * ws[k];
    out[r] = 1.f / (1.f + __expf(-s));
}

// ================= host-side orchestration ===================================
static Gather3 mkg(const float* p0, const int* i0, int w0, int l0,
                   const float* p1 = nullptr, const int* i1 = nullptr, int w1 = 0, int l1 = 1,
                   const float* p2 = nullptr, const int* i2 = nullptr, int w2 = 0, int l2 = 1)
{
    Gather3 g;
    g.ptr[0] = p0; g.idx[0] = i0; g.width[0] = w0; g.ld[0] = l0;
    g.ptr[1] = p1; g.idx[1] = i1; g.width[1] = w1; g.ld[1] = l1;
    g.ptr[2] = p2; g.idx[2] = i2; g.width[2] = w2; g.ld[2] = l2;
    return g;
}

static char* carve(char*& cur, size_t bytes)
{
    char* p = cur;
    cur += (bytes + 255) & ~(size_t)255;
    return p;
}

static void gemm(hipStream_t st, Gather3 g, const float* W, int ldw, const float* bias,
                 float* Y, int ldy, int ycol0, int M, int N, int K)
{
    dim3 grid((M + TM - 1) / TM, (N + TN - 1) / TN);
    wmma_gemm_gather<<<grid, 256, 0, st>>>(W, ldw, bias, Y, ldy, ycol0, M, N, K, g);
}

static void bn_lrelu(hipStream_t st, float* Y, int ld, int M,
                     const float* gamma, const float* beta, float* stats)
{
    fillk<<<1, 128, 0, st>>>(stats, 128, 0.f);
    bn_stats64<<<256, 256, 0, st>>>(Y, ld, M, stats);
    long long total = (long long)M * 64;
    act_kernel<<<(unsigned)((total + 255) / 256), 256, 0, st>>>(
        Y, ld, M, 64, 1, stats, gamma, beta, 1.f / (float)M);
}

static void act_only(hipStream_t st, float* Y, int ld, int M, int C, int mode)
{
    long long total = (long long)M * C;
    act_kernel<<<(unsigned)((total + 255) / 256), 256, 0, st>>>(
        Y, ld, M, C, mode, nullptr, nullptr, nullptr, 0.f);
}

static void scatter_mean(hipStream_t st, const float* msg, const int* dst, int E,
                         float* agg, float* cnt)
{
    fillk<<<(unsigned)(((long long)N_NODES * 64 + 255) / 256), 256, 0, st>>>(agg, (long long)N_NODES * 64, 0.f);
    fillk<<<(N_NODES + 255) / 256, 256, 0, st>>>(cnt, N_NODES, 0.f);
    scatter_add64<<<(unsigned)(((long long)E * 64 + 255) / 256), 256, 0, st>>>(msg, dst, E, agg);
    scatter_cnt<<<(E + 255) / 256, 256, 0, st>>>(dst, E, cnt);
    divide64<<<(unsigned)(((long long)N_NODES * 64 + 255) / 256), 256, 0, st>>>(agg, cnt, N_NODES);
}

extern "C" void kernel_launch(void* const* d_in, const int* in_sizes, int n_in,
                              void* d_out, int out_size, void* d_ws, size_t ws_size,
                              hipStream_t stream)
{
    const float* x   = (const float*)d_in[0];
    const int*   adj = (const int*)d_in[1];
    const float* ef  = (const float*)d_in[2];
    const int*   col = (const int*)d_in[3];
    const int* adj_src = adj;            const int* adj_dst = adj + N_ADJ;
    const int* col_src = col;            const int* col_dst = col + N_COL;

    // ---- params: jax dict tree-flatten order (alphabetical keys) ----
    int pi = 4;
    auto nxt = [&]() { return (const float*)d_in[pi++]; };

    const float *coll_gate_b[DEPTH], *coll_gate_w[DEPTH];
    const float *cm_be[DEPTH][2], *cm_g[DEPTH][2], *cm_b[DEPTH][2], *cm_w[DEPTH][2];
    for (int i = 0; i < DEPTH; ++i) {
        coll_gate_b[i] = nxt(); coll_gate_w[i] = nxt();
        for (int l = 0; l < 2; ++l) {
            cm_be[i][l] = nxt(); cm_g[i][l] = nxt();
            cm_b[i][l]  = nxt(); cm_w[i][l] = nxt();
        }
    }
    const float* flin_b = nxt(); const float* flin_w = nxt();
    const float *fm_b[4], *fm_w[4];
    for (int l = 0; l < 4; ++l) { fm_b[l] = nxt(); fm_w[l] = nxt(); }
    const float *ge_be[DEPTH][2], *ge_g[DEPTH][2], *ge_b[DEPTH][2], *ge_w[DEPTH][2];
    const float *gn_be[DEPTH][2], *gn_g[DEPTH][2], *gn_b[DEPTH][2], *gn_w[DEPTH][2];
    for (int i = 0; i < DEPTH; ++i) {
        for (int l = 0; l < 2; ++l) {
            ge_be[i][l] = nxt(); ge_g[i][l] = nxt();
            ge_b[i][l]  = nxt(); ge_w[i][l] = nxt();
        }
        for (int l = 0; l < 2; ++l) {
            gn_be[i][l] = nxt(); gn_g[i][l] = nxt();
            gn_b[i][l]  = nxt(); gn_w[i][l] = nxt();
        }
    }
    const float *in_be[2], *in_g[2], *in_b[2], *in_w[2];
    for (int l = 0; l < 2; ++l) {
        in_be[l] = nxt(); in_g[l] = nxt();
        in_b[l]  = nxt(); in_w[l] = nxt();
    }

    // ---- workspace carve ----
    char* cur = (char*)d_ws;
    float* FEATS = (float*)carve(cur, (size_t)N_NODES * 448 * 4); // concat buffer (fits L2)
    float* b2a   = (float*)carve(cur, (size_t)N_NODES * 64 * 4);
    float* b2b   = (float*)carve(cur, (size_t)N_NODES * 64 * 4);
    float* eY1   = (float*)carve(cur, (size_t)N_ADJ * 64 * 4);
    float* eY2   = (float*)carve(cur, (size_t)N_ADJ * 64 * 4);
    float* cY1   = (float*)carve(cur, (size_t)N_COL * 64 * 4);
    float* cY2   = (float*)carve(cur, (size_t)N_COL * 64 * 4);
    float* agg   = (float*)carve(cur, (size_t)N_NODES * 64 * 4);
    float* cnt   = (float*)carve(cur, (size_t)N_NODES * 4);
    float* agg2  = (float*)carve(cur, (size_t)N_NODES * 64 * 4);
    float* cnt2  = (float*)carve(cur, (size_t)N_NODES * 4);
    float* stats = (float*)carve(cur, 128 * 4);
    float* nY    = (float*)carve(cur, (size_t)N_NODES * 64 * 4);
    float* nY2   = (float*)carve(cur, (size_t)N_NODES * 64 * 4);
    float* h0    = (float*)carve(cur, (size_t)N_NODES * 256 * 4);
    float* h1    = (float*)carve(cur, (size_t)N_NODES * 128 * 4);
    float* h2    = (float*)carve(cur, (size_t)N_NODES * 64 * 4);
    float* h3    = (float*)carve(cur, (size_t)N_NODES * 64 * 4);

    // ---- init MLP: 46 -> 64 -> 64 (BN + lrelu), second layer into FEATS[:,0:64] ----
    gemm(stream, mkg(x, nullptr, NODE_IN, NODE_IN), in_w[0], MID, in_b[0],
         nY, MID, 0, N_NODES, MID, NODE_IN);
    bn_lrelu(stream, nY, MID, N_NODES, in_g[0], in_be[0], stats);
    gemm(stream, mkg(nY, nullptr, MID, MID), in_w[1], MID, in_b[1],
         FEATS, 448, 0, N_NODES, MID, MID);
    bn_lrelu(stream, FEATS, 448, N_NODES, in_g[1], in_be[1], stats);

    // b2 = ones (width 1 initially)
    fillk<<<(N_NODES + 255) / 256, 256, 0, stream>>>(b2a, N_NODES, 1.f);
    float* b2 = b2a; float* b2o = b2b;
    int in2 = 1;

    for (int i = 0; i < DEPTH; ++i) {
        const float* b1 = FEATS + (size_t)i * 64; // current b1 slice, ld 448

        // ---- graph conv: edge MLP (gather [x_src | x_dst | e_feat], K=135) ----
        gemm(stream,
             mkg(b1, adj_src, MID, 448, b1, adj_dst, MID, 448, ef, nullptr, EDGE_F, EDGE_F),
             ge_w[i][0], MID, ge_b[i][0], eY1, MID, 0, N_ADJ, MID, 2 * MID + EDGE_F);
        bn_lrelu(stream, eY1, MID, N_ADJ, ge_g[i][0], ge_be[i][0], stats);
        gemm(stream, mkg(eY1, nullptr, MID, MID), ge_w[i][1], MID, ge_b[i][1],
             eY2, MID, 0, N_ADJ, MID, MID);
        bn_lrelu(stream, eY2, MID, N_ADJ, ge_g[i][1], ge_be[i][1], stats);

        scatter_mean(stream, eY2, adj_dst, N_ADJ, agg, cnt);

        // ---- node MLP: [b1 | agg], K=128 ----
        gemm(stream, mkg(b1, nullptr, MID, 448, agg, nullptr, MID, MID),
             gn_w[i][0], MID, gn_b[i][0], nY, MID, 0, N_NODES, MID, 2 * MID);
        bn_lrelu(stream, nY, MID, N_NODES, gn_g[i][0], gn_be[i][0], stats);
        gemm(stream, mkg(nY, nullptr, MID, MID), gn_w[i][1], MID, gn_b[i][1],
             nY2, MID, 0, N_NODES, MID, MID);
        bn_lrelu(stream, nY2, MID, N_NODES, gn_g[i][1], gn_be[i][1], stats);

        // ---- collision conv: msg MLP on [b2_src | b2_dst], K=2*in2 ----
        gemm(stream, mkg(b2, col_src, in2, in2, b2, col_dst, in2, in2),
             cm_w[i][0], MID, cm_b[i][0], cY1, MID, 0, N_COL, MID, 2 * in2);
        bn_lrelu(stream, cY1, MID, N_COL, cm_g[i][0], cm_be[i][0], stats);
        gemm(stream, mkg(cY1, nullptr, MID, MID), cm_w[i][1], MID, cm_b[i][1],
             cY2, MID, 0, N_COL, MID, MID);
        bn_lrelu(stream, cY2, MID, N_COL, cm_g[i][1], cm_be[i][1], stats);

        scatter_mean(stream, cY2, col_dst, N_COL, agg2, cnt2);

        // gate: sigmoid(linear([b2 | agg2])), K = in2 + 64
        gemm(stream, mkg(b2, nullptr, in2, in2, agg2, nullptr, MID, MID),
             coll_gate_w[i], MID, coll_gate_b[i], b2o, MID, 0, N_NODES, MID, in2 + MID);
        act_only(stream, b2o, MID, N_NODES, MID, 2 /*sigmoid*/);
        { float* t = b2; b2 = b2o; b2o = t; }
        in2 = MID;

        // ---- gating multiply + residual skip, store into FEATS[:, (i+1)*64] ----
        const float* resid = (i >= 2) ? (FEATS + (size_t)(i - 2) * 64) : nullptr;
        gate_res<<<(unsigned)(((long long)N_NODES * 64 + 255) / 256), 256, 0, stream>>>(
            nY2, b2, resid, 448, FEATS + (size_t)(i + 1) * 64, 448, N_NODES);
    }

    // ---- final MLP: 448 -> 256 -> 128 -> 64 -> 64 (lrelu, no BN) ----
    gemm(stream, mkg(FEATS, nullptr, 448, 448), fm_w[0], 256, fm_b[0],
         h0, 256, 0, N_NODES, 256, 448);
    act_only(stream, h0, 256, N_NODES, 256, 0);
    gemm(stream, mkg(h0, nullptr, 256, 256), fm_w[1], 128, fm_b[1],
         h1, 128, 0, N_NODES, 128, 256);
    act_only(stream, h1, 128, N_NODES, 128, 0);
    gemm(stream, mkg(h1, nullptr, 128, 128), fm_w[2], MID, fm_b[2],
         h2, MID, 0, N_NODES, MID, 128);
    act_only(stream, h2, MID, N_NODES, MID, 0);
    gemm(stream, mkg(h2, nullptr, MID, MID), fm_w[3], MID, fm_b[3],
         h3, MID, 0, N_NODES, MID, MID);
    act_only(stream, h3, MID, N_NODES, MID, 0);

    // ---- final head: 64 -> 1 + sigmoid ----
    final_head<<<(N_NODES + 255) / 256, 256, 0, stream>>>(
        h3, flin_w, flin_b, (float*)d_out, N_NODES);
}
